// RGN_18803366822570
// MI455X (gfx1250) — matrix operand order repaired and checked by reading, not compiled, not run on hardware
//
#include <hip/hip_runtime.h>
#include <hip/hip_bf16.h>
#include <stdint.h>

// ---------------- shapes ----------------
#define TT   512
#define BSZ  32
#define DIN  42
#define DPAD 64
#define HH   800
#define G4   3200   // 4*H
#define H2   1600   // 2*H
#define ROWS (TT*BSZ)   // 16384
#define KOUT 20
#define JT   16         // hidden tile per block in scan kernel

typedef __bf16 bf16;
typedef __attribute__((ext_vector_type(16))) __bf16 v16bf;
typedef __attribute__((ext_vector_type(8)))  float  v8f;

typedef int v4i __attribute__((vector_size(16)));
typedef __attribute__((address_space(1))) v4i* gv4ip;
typedef __attribute__((address_space(3))) v4i* lv4ip;

#if __has_builtin(__builtin_amdgcn_global_load_async_to_lds_b128)
#define HAVE_ASYNC_LDS 1
#endif
#if __has_builtin(__builtin_amdgcn_s_wait_asynccnt)
#define HAVE_WAIT_ASYNC 1
#endif

union FragU { v16bf v; unsigned int u[8]; };

// A fragment (16x32 bf16), tile in LDS row-major stride 32.
__device__ __forceinline__ v16bf load_a_frag(const bf16* lds, int mOff, int lane) {
  FragU r;
  int m  = mOff + (lane & 15);
  int kb = (lane >> 4) << 3;               // 0 or 8
  #pragma unroll
  for (int v = 0; v < 8; ++v) {
    int k = ((v >= 4) ? 16 : 0) + kb + ((v & 3) << 1);
    r.u[v] = *(const unsigned int*)(lds + m * 32 + k);
  }
  return r.v;
}

// B fragment (32x16 bf16): lane L holds col n=nOff+L%16; element s holds k=(s>=8?16:0)+8*(L/16)+(s&7).
__device__ __forceinline__ v16bf load_b_frag(const bf16* lds, int ldb, int nOff, int lane) {
  FragU r;
  int n  = nOff + (lane & 15);
  int kb = (lane >> 4) << 3;
  #pragma unroll
  for (int v = 0; v < 16; ++v)
    r.v[v] = lds[(((v >= 8) ? 16 : 0) + kb + (v & 7)) * ldb + n];
  return r.v;
}

__device__ __forceinline__ v8f wmma_bf16(v16bf a, v16bf b, v8f c) {
  return __builtin_amdgcn_wmma_f32_16x16x32_bf16(false, a, false, b, (short)0, c, false, false);
}

__device__ __forceinline__ void async_copy16(const bf16* gsrc, bf16* ldst) {
#if defined(HAVE_ASYNC_LDS)
  __builtin_amdgcn_global_load_async_to_lds_b128((gv4ip)gsrc, (lv4ip)ldst, 0, 0);
#else
  *(uint4*)ldst = *(const uint4*)gsrc;
#endif
}

__device__ __forceinline__ void wait_async0() {
#if defined(HAVE_ASYNC_LDS) && defined(HAVE_WAIT_ASYNC)
  __builtin_amdgcn_s_wait_asynccnt(0);
#endif
}

// ---------------- conversion kernels ----------------

__global__ void convert_x_kernel(const float* __restrict__ x, bf16* __restrict__ out) {
  int idx = blockIdx.x * blockDim.x + threadIdx.x;
  if (idx >= ROWS * DPAD) return;
  int d   = idx & (DPAD - 1);
  int row = idx >> 6;
  int t = row >> 5, b = row & 31;
  float v = (d < DIN) ? x[((size_t)b * TT + t) * DIN + d] : 0.0f;
  out[idx] = (bf16)v;
}

__global__ void conv_wT_kernel(const float* __restrict__ in, bf16* __restrict__ out,
                               int J, int Kd, int Kp) {
  int idx = blockIdx.x * blockDim.x + threadIdx.x;
  if (idx >= Kp * J) return;
  int k = idx / J, j = idx - k * J;
  out[idx] = (bf16)((k < Kd) ? in[(size_t)j * Kd + k] : 0.0f);
}

__global__ void zero_u32_kernel(unsigned int* p, int n) {
  int i = blockIdx.x * blockDim.x + threadIdx.x;
  if (i < n) p[i] = 0u;
}

// ---------------- big GEMM: C[M][N]=A[M][lda(K)] @ B[K][N] + bias, bf16 out ----------------
__global__ __launch_bounds__(256) void gemm_bias_kernel(
    const bf16* __restrict__ A, int lda,
    const bf16* __restrict__ B,
    const float* __restrict__ bias,
    bf16* __restrict__ C,
    int N, int K) {
  __shared__ __align__(16) bf16 As[128 * 32];
  __shared__ __align__(16) bf16 Bs[32 * 128];
  int tid  = threadIdx.x;
  int lane = tid & 31;
  int wave = tid >> 5;
  int wm = wave >> 1;           // 0..3 -> 32 rows each
  int wn = wave & 1;            // 0..1 -> 64 cols each
  int bm = blockIdx.y * 128;
  int bn = blockIdx.x * 128;

  v8f vz = {0.f,0.f,0.f,0.f,0.f,0.f,0.f,0.f};
  v8f acc[2][4];
  #pragma unroll
  for (int i = 0; i < 2; ++i)
    #pragma unroll
    for (int j = 0; j < 4; ++j) acc[i][j] = vz;

  for (int kc = 0; kc < K; kc += 32) {
    #pragma unroll
    for (int rep = 0; rep < 2; ++rep) {           // A tile 128x32
      int q = tid + rep * 256;
      int m = q >> 2, seg = q & 3;
      *(uint4*)&As[m * 32 + seg * 8] =
          *(const uint4*)(A + (size_t)(bm + m) * lda + kc + seg * 8);
    }
    #pragma unroll
    for (int rep = 0; rep < 2; ++rep) {           // B tile 32x128
      int q = tid + rep * 256;
      int kr = q >> 4, seg = q & 15;
      *(uint4*)&Bs[kr * 128 + seg * 8] =
          *(const uint4*)(B + (size_t)(kc + kr) * N + bn + seg * 8);
    }
    __syncthreads();
    v16bf af[2], bfv[4];
    af[0] = load_a_frag(As, wm * 32 + 0,  lane);
    af[1] = load_a_frag(As, wm * 32 + 16, lane);
    #pragma unroll
    for (int j = 0; j < 4; ++j) bfv[j] = load_b_frag(Bs, 128, wn * 64 + j * 16, lane);
    #pragma unroll
    for (int i = 0; i < 2; ++i)
      #pragma unroll
      for (int j = 0; j < 4; ++j) acc[i][j] = wmma_bf16(af[i], bfv[j], acc[i][j]);
    __syncthreads();
  }

  #pragma unroll
  for (int i = 0; i < 2; ++i) {
    int mBase = bm + wm * 32 + i * 16 + ((lane < 16) ? 0 : 8);
    #pragma unroll
    for (int j = 0; j < 4; ++j) {
      int n = bn + wn * 64 + j * 16 + (lane & 15);
      float bv = bias ? bias[n] : 0.0f;
      #pragma unroll
      for (int r = 0; r < 8; ++r)
        C[(size_t)(mBase + r) * N + n] = (bf16)(acc[i][j][r] + bv);
    }
  }
}

// ---------------- persistent bidirectional LSTM scan ----------------
// grid (HH/JT=50, 2): x = hidden tile (JT cols), y = direction. block 128 = 4 waves (one per gate).
// whh gate-slice stays resident in LDS for all 512 steps. A fragments are loaded directly
// from L2 (barrier-free K-loop); the xg gate tile is prefetched to LDS with async
// global->LDS copies overlapped with the whole recurrent GEMM.
#define SCAN_LDS_BYTES (4*HH*JT*2 /*B*/ + 4*32*JT*2 /*xg tile*/ + 4*32*JT*4 /*gates*/)

__global__ __launch_bounds__(128) void lstm_scan_kernel(
    const bf16* __restrict__ xg_f, const bf16* __restrict__ xg_b,     // [ROWS][G4]
    const bf16* __restrict__ whhT_f, const bf16* __restrict__ whhT_b, // [HH][G4]
    bf16* __restrict__ h_all,          // [ROWS][H2]
    float* __restrict__ c_buf,         // [2][32][HH]
    unsigned int* __restrict__ bar) {
  extern __shared__ __align__(16) char smem[];
  bf16*  Bsl = (bf16*)smem;                               // [4][HH][JT]
  bf16*  Gx  = (bf16*)(smem + 4*HH*JT*2);                 // [4][32][JT] xg tile
  float* G   = (float*)(smem + 4*HH*JT*2 + 4*32*JT*2);    // [4][32][JT] gates

  int tid  = threadIdx.x;
  int lane = tid & 31;
  int w    = tid >> 5;                  // gate: 0=i 1=f 2=g 3=o
  int dir  = blockIdx.y;
  int j0   = blockIdx.x * JT;
  int dirOff = dir * HH;
  const bf16* xg   = (dir == 0) ? xg_f   : xg_b;
  const bf16* whhT = (dir == 0) ? whhT_f : whhT_b;
  unsigned int nblocks = gridDim.x * gridDim.y;

  // ---- preload whh gate-slice into LDS (resident for whole scan) ----
  for (int it = 0; it < (4 * HH * JT) / (128 * 8); ++it) {   // 50 iters
    int q = it * 128 + tid;            // 0..6399, 8 elems each
    int wg  = q / (HH * 2);
    int rem = q - wg * (HH * 2);
    int k   = rem >> 1, seg = rem & 1;
    *(uint4*)&Bsl[((size_t)wg * HH + k) * JT + seg * 8] =
        *(const uint4*)(whhT + (size_t)k * G4 + wg * HH + j0 + seg * 8);
  }
  __syncthreads();

  v8f vz = {0.f,0.f,0.f,0.f,0.f,0.f,0.f,0.f};
  int m  = lane & 15;
  int kb = (lane >> 4) << 3;            // 0 or 8: A-fragment per-lane K base

  for (int tt = 0; tt < TT; ++tt) {
    int t     = (dir == 0) ? tt : (TT - 1 - tt);
    int tprev = (dir == 0) ? t - 1 : t + 1;
    bool first = (tt == 0);

    // ---- async prefetch of xg gate tile (overlaps the whole recurrent GEMM) ----
    #pragma unroll
    for (int r = 0; r < 2; ++r) {
      int q = tid * 2 + r;              // 0..255
      int wq = q >> 6, bq = (q >> 1) & 31, seg = q & 1;
      async_copy16(xg + (size_t)(t * BSZ + bq) * G4 + wq * HH + j0 + seg * 8,
                   &Gx[(wq * 32 + bq) * JT + seg * 8]);
    }

    v8f acc0 = vz, acc1 = vz;

    if (!first) {
      // barrier-free K-loop: A fragments straight from L2-resident h_prev
      const bf16* hp   = h_all + (size_t)tprev * BSZ * H2 + dirOff;
      const bf16* rowA = hp + (size_t)m * H2 + kb;          // batch rows 0..15
      const bf16* rowB = hp + (size_t)(m + 16) * H2 + kb;   // batch rows 16..31
      const bf16* bbase = Bsl + (size_t)w * HH * JT + (lane & 15);
      for (int i = 0; i < HH / 32; ++i) {                   // 25 chunks
        FragU a0, a1, bf_;
        *(uint4*)&a0.u[0] = *(const uint4*)(rowA + i * 32);
        *(uint4*)&a0.u[4] = *(const uint4*)(rowA + i * 32 + 16);
        *(uint4*)&a1.u[0] = *(const uint4*)(rowB + i * 32);
        *(uint4*)&a1.u[4] = *(const uint4*)(rowB + i * 32 + 16);
        const bf16* bp = bbase + (size_t)i * 32 * JT;
        #pragma unroll
        for (int v = 0; v < 16; ++v) bf_.v[v] = bp[(((v >= 8) ? 16 : 0) + kb + (v & 7)) * JT];
        acc0 = wmma_bf16(a0.v, bf_.v, acc0);
        acc1 = wmma_bf16(a1.v, bf_.v, acc1);
      }
    }

    wait_async0();        // xg tile landed (per-wave ASYNCcnt)
    __syncthreads();

    // gates = acc + xg -> LDS
    #pragma unroll
    for (int i = 0; i < 2; ++i) {
      int bm = i * 16 + ((lane < 16) ? 0 : 8);
      int nl = lane & 15;
      const v8f* ap = (i == 0) ? &acc0 : &acc1;
      #pragma unroll
      for (int r = 0; r < 8; ++r) {
        int b = bm + r;
        float xv = (float)Gx[(w * 32 + b) * JT + nl];
        G[(w * 32 + b) * JT + nl] = (*ap)[r] + xv;
      }
    }
    __syncthreads();

    // cell update: 32*JT = 512 elems, 4 per thread
    #pragma unroll
    for (int e = 0; e < 4; ++e) {
      int idx = tid * 4 + e;
      int b = idx >> 4, jl = idx & 15;
      float gi = G[(0 * 32 + b) * JT + jl];
      float gf = G[(1 * 32 + b) * JT + jl];
      float gg = G[(2 * 32 + b) * JT + jl];
      float go = G[(3 * 32 + b) * JT + jl];
      float* cp = c_buf + (size_t)dir * BSZ * HH + b * HH + j0 + jl;
      float c = first ? 0.0f : *cp;
      float si = 1.0f / (1.0f + __expf(-gi));
      float sf = 1.0f / (1.0f + __expf(-gf));
      float so = 1.0f / (1.0f + __expf(-go));
      c = sf * c + si * tanhf(gg);
      float h = so * tanhf(c);
      *cp = c;
      h_all[(size_t)(t * BSZ + b) * H2 + dirOff + j0 + jl] = (bf16)h;
    }

    // ---- device-wide step barrier ----
    __threadfence();
    if (tid == 0) {
      __hip_atomic_fetch_add(bar, 1u, __ATOMIC_ACQ_REL, __HIP_MEMORY_SCOPE_AGENT);
      unsigned int target = (unsigned int)(tt + 1) * nblocks;
      while (__hip_atomic_load(bar, __ATOMIC_ACQUIRE, __HIP_MEMORY_SCOPE_AGENT) < target)
        __builtin_amdgcn_s_sleep(1);
    }
    __syncthreads();
    __threadfence();
  }
}

// ---------------- head ----------------
__global__ void logits_kernel(const bf16* __restrict__ h2, const float* __restrict__ lw,
                              const float* __restrict__ lb, float* __restrict__ out) {
  int idx = blockIdx.x * blockDim.x + threadIdx.x;
  if (idx >= ROWS * KOUT) return;
  int j = idx % KOUT, row = idx / KOUT;
  const bf16* hp = h2 + (size_t)row * H2;
  const float* wp = lw + (size_t)j * H2;
  float acc = 0.0f;
  for (int k = 0; k < H2; ++k) acc += (float)hp[k] * wp[k];
  out[idx] = acc + lb[j];
}

__global__ void softang_kernel(const float* __restrict__ logits,
                               const float* __restrict__ alphabet,
                               float* __restrict__ ang) {
  int t = blockIdx.x, b = threadIdx.x;
  float sx[3] = {0.f, 0.f, 0.f}, sy[3] = {0.f, 0.f, 0.f};
  #pragma unroll
  for (int k = 0; k < KOUT; ++k) {
    float v = logits[((size_t)t * BSZ + b) * KOUT + k];
    float m = v;
    #pragma unroll
    for (int off = 16; off; off >>= 1) m = fmaxf(m, __shfl_xor(m, off, 32));
    float e = __expf(v - m);
    float s = e;
    #pragma unroll
    for (int off = 16; off; off >>= 1) s += __shfl_xor(s, off, 32);
    float p = e / s;
    #pragma unroll
    for (int i = 0; i < 3; ++i) {
      float a = alphabet[k * 3 + i];
      sy[i] += p * __sinf(a);
      sx[i] += p * __cosf(a);
    }
  }
  #pragma unroll
  for (int i = 0; i < 3; ++i)
    ang[((size_t)t * BSZ + b) * 3 + i] = atan2f(sy[i], sx[i]);
}

__global__ void coords_kernel(const float* __restrict__ ang, float* __restrict__ out) {
  int b = threadIdx.x;
  const float LENS[3] = {145.801f, 152.326f, 132.868f};
  const float ANGS[3] = {2.124f, 1.941f, 2.028f};
  float ax = 0.f, ay = 0.f, az = 1.f;
  float bx = 0.f, by = 1.f, bz = 1.f;
  float cx = 1.f, cy = 1.f, cz = 1.f;
  out[(size_t)(0 * BSZ + b) * 3 + 0] = ax; out[(size_t)(0 * BSZ + b) * 3 + 1] = ay; out[(size_t)(0 * BSZ + b) * 3 + 2] = az;
  out[(size_t)(1 * BSZ + b) * 3 + 0] = bx; out[(size_t)(1 * BSZ + b) * 3 + 1] = by; out[(size_t)(1 * BSZ + b) * 3 + 2] = bz;
  out[(size_t)(2 * BSZ + b) * 3 + 0] = cx; out[(size_t)(2 * BSZ + b) * 3 + 1] = cy; out[(size_t)(2 * BSZ + b) * 3 + 2] = cz;
  for (int t = 0; t < TT; ++t) {
    #pragma unroll
    for (int i = 0; i < 3; ++i) {
      float r = LENS[i], th = ANGS[i];
      float p = ang[((size_t)t * BSZ + b) * 3 + i];
      float d2x = -r * __cosf(th);
      float d2y =  r * __cosf(p) * __sinf(th);
      float d2z =  r * __sinf(p) * __sinf(th);
      float bcx = cx - bx, bcy = cy - by, bcz = cz - bz;
      float il = rsqrtf(bcx * bcx + bcy * bcy + bcz * bcz);
      bcx *= il; bcy *= il; bcz *= il;
      float abx = bx - ax, aby = by - ay, abz = bz - az;
      float nx = aby * bcz - abz * bcy;
      float ny = abz * bcx - abx * bcz;
      float nz = abx * bcy - aby * bcx;
      il = rsqrtf(nx * nx + ny * ny + nz * nz);
      nx *= il; ny *= il; nz *= il;
      float mx = ny * bcz - nz * bcy;
      float my = nz * bcx - nx * bcz;
      float mz = nx * bcy - ny * bcx;
      float dx = bcx * d2x + mx * d2y + nx * d2z + cx;
      float dy = bcy * d2x + my * d2y + ny * d2z + cy;
      float dz = bcz * d2x + mz * d2y + nz * d2z + cz;
      int row = 3 + t * 3 + i;
      out[(size_t)(row * BSZ + b) * 3 + 0] = dx;
      out[(size_t)(row * BSZ + b) * 3 + 1] = dy;
      out[(size_t)(row * BSZ + b) * 3 + 2] = dz;
      ax = bx; ay = by; az = bz;
      bx = cx; by = cy; bz = cz;
      cx = dx; cy = dy; cz = dz;
    }
  }
}

// ---------------- host ----------------
extern "C" void kernel_launch(void* const* d_in, const int* in_sizes, int n_in,
                              void* d_out, int out_size, void* d_ws, size_t ws_size,
                              hipStream_t stream) {
  const float* x      = (const float*)d_in[0];
  const float* wih0f  = (const float*)d_in[1];
  const float* whh0f  = (const float*)d_in[2];
  const float* b0f    = (const float*)d_in[3];
  const float* wih0b  = (const float*)d_in[4];
  const float* whh0b  = (const float*)d_in[5];
  const float* b0b    = (const float*)d_in[6];
  const float* wih1f  = (const float*)d_in[7];
  const float* whh1f  = (const float*)d_in[8];
  const float* b1f    = (const float*)d_in[9];
  const float* wih1b  = (const float*)d_in[10];
  const float* whh1b  = (const float*)d_in[11];
  const float* b1b    = (const float*)d_in[12];
  const float* lin_w  = (const float*)d_in[13];
  const float* lin_b  = (const float*)d_in[14];
  const float* alpha  = (const float*)d_in[15];

  char* ws = (char*)d_ws;
  size_t off = 0;
  auto alloc = [&](size_t bytes) { size_t o = off; off += (bytes + 255) & ~(size_t)255; return o; };

  bf16* xbf     = (bf16*)(ws + alloc((size_t)ROWS * DPAD * 2));
  bf16* wih0fT  = (bf16*)(ws + alloc((size_t)DPAD * G4 * 2));
  bf16* wih0bT  = (bf16*)(ws + alloc((size_t)DPAD * G4 * 2));
  bf16* wih1fT  = (bf16*)(ws + alloc((size_t)H2 * G4 * 2));
  bf16* wih1bT  = (bf16*)(ws + alloc((size_t)H2 * G4 * 2));
  bf16* whh0fT  = (bf16*)(ws + alloc((size_t)HH * G4 * 2));
  bf16* whh0bT  = (bf16*)(ws + alloc((size_t)HH * G4 * 2));
  bf16* whh1fT  = (bf16*)(ws + alloc((size_t)HH * G4 * 2));
  bf16* whh1bT  = (bf16*)(ws + alloc((size_t)HH * G4 * 2));
  bf16* xg_f    = (bf16*)(ws + alloc((size_t)ROWS * G4 * 2));
  bf16* xg_b    = (bf16*)(ws + alloc((size_t)ROWS * G4 * 2));
  bf16* h1      = (bf16*)(ws + alloc((size_t)ROWS * H2 * 2));
  bf16* h2      = (bf16*)(ws + alloc((size_t)ROWS * H2 * 2));
  float* cbuf   = (float*)(ws + alloc((size_t)2 * BSZ * HH * 4));
  float* logits = (float*)(ws + alloc((size_t)ROWS * KOUT * 4));
  float* angb   = (float*)(ws + alloc((size_t)ROWS * 3 * 4));
  unsigned int* bar = (unsigned int*)(ws + alloc(256));

  (void)hipFuncSetAttribute((const void*)lstm_scan_kernel,
                            hipFuncAttributeMaxDynamicSharedMemorySize, SCAN_LDS_BYTES);

  // conversions
  convert_x_kernel<<<(ROWS * DPAD + 255) / 256, 256, 0, stream>>>(x, xbf);
  conv_wT_kernel<<<(DPAD * G4 + 255) / 256, 256, 0, stream>>>(wih0f, wih0fT, G4, DIN, DPAD);
  conv_wT_kernel<<<(DPAD * G4 + 255) / 256, 256, 0, stream>>>(wih0b, wih0bT, G4, DIN, DPAD);
  conv_wT_kernel<<<(H2 * G4 + 255) / 256, 256, 0, stream>>>(wih1f, wih1fT, G4, H2, H2);
  conv_wT_kernel<<<(H2 * G4 + 255) / 256, 256, 0, stream>>>(wih1b, wih1bT, G4, H2, H2);
  conv_wT_kernel<<<(HH * G4 + 255) / 256, 256, 0, stream>>>(whh0f, whh0fT, G4, HH, HH);
  conv_wT_kernel<<<(HH * G4 + 255) / 256, 256, 0, stream>>>(whh0b, whh0bT, G4, HH, HH);
  conv_wT_kernel<<<(HH * G4 + 255) / 256, 256, 0, stream>>>(whh1f, whh1fT, G4, HH, HH);
  conv_wT_kernel<<<(HH * G4 + 255) / 256, 256, 0, stream>>>(whh1b, whh1bT, G4, HH, HH);

  dim3 ggrid(G4 / 128, ROWS / 128);
  // layer-0 input projections
  gemm_bias_kernel<<<ggrid, 256, 0, stream>>>(xbf, DPAD, wih0fT, b0f, xg_f, G4, DPAD);
  gemm_bias_kernel<<<ggrid, 256, 0, stream>>>(xbf, DPAD, wih0bT, b0b, xg_b, G4, DPAD);

  // layer-0 persistent scan (both directions concurrently)
  zero_u32_kernel<<<1, 64, 0, stream>>>(bar, 64);
  lstm_scan_kernel<<<dim3(HH / JT, 2), 128, SCAN_LDS_BYTES, stream>>>(
      xg_f, xg_b, whh0fT, whh0bT, h1, cbuf, bar);

  // layer-1 input projections from h1
  gemm_bias_kernel<<<ggrid, 256, 0, stream>>>(h1, H2, wih1fT, b1f, xg_f, G4, H2);
  gemm_bias_kernel<<<ggrid, 256, 0, stream>>>(h1, H2, wih1bT, b1b, xg_b, G4, H2);

  // layer-1 persistent scan
  zero_u32_kernel<<<1, 64, 0, stream>>>(bar, 64);
  lstm_scan_kernel<<<dim3(HH / JT, 2), 128, SCAN_LDS_BYTES, stream>>>(
      xg_f, xg_b, whh1fT, whh1bT, h2, cbuf, bar);

  // head
  logits_kernel<<<(ROWS * KOUT + 255) / 256, 256, 0, stream>>>(h2, lin_w, lin_b, logits);
  softang_kernel<<<TT, 32, 0, stream>>>(logits, alpha, angb);
  coords_kernel<<<1, 32, 0, stream>>>(angb, (float*)d_out);

  (void)in_sizes; (void)n_in; (void)out_size; (void)ws_size;
}
